// Decoder_31550829757123
// MI455X (gfx1250) — matrix operand order repaired and checked by reading, not compiled, not run on hardware
//
#include <hip/hip_runtime.h>
#include <math.h>

typedef __attribute__((ext_vector_type(16))) _Float16 v16h;
typedef __attribute__((ext_vector_type(8)))  _Float16 v8h;
typedef __attribute__((ext_vector_type(8)))  float    v8f;

// ---- problem constants (match reference) ----
#define LAYERS  2
#define HDIM    1024
#define VOCAB   50257
#define NSTEPS  128
#define G3H     (3 * HDIM)           // 3072
#define TILES_LOGITS 3144            // 393 blocks * 8 waves, covers ceil(50257/16)=3142
#define VPAD    (TILES_LOGITS * 16)  // 50304, zero-padded rows for full-EXEC WMMA
#define NBLK_P  (TILES_LOGITS / 8)   // 393 logits blocks -> softmax partials
#define NBLK_V  ((VOCAB + 255) / 256) // 197 blocks for logp write

#define NEG_INF (-__builtin_huge_valf())

// =====================================================================
// fp32 -> f16 conversion with zero padding (run once per call; puts the
// whole weight working set (~128MB f16) into L2-resident form)
// =====================================================================
__global__ void dec_cvt_f16(const float* __restrict__ src, _Float16* __restrict__ dst,
                            long n_src, long n_total) {
    long i = ((long)blockIdx.x * blockDim.x + threadIdx.x) * 4;
    if (i >= n_total) return;
#pragma unroll
    for (int k = 0; k < 4; ++k) {
        long j = i + k;
        if (j < n_total) dst[j] = (j < n_src) ? (_Float16)src[j] : (_Float16)0.0f;
    }
}

__global__ void dec_init_state(const float* __restrict__ hidden,
                               float* __restrict__ hF32, _Float16* __restrict__ hF16,
                               const int* __restrict__ sos, int* __restrict__ word) {
    int i = blockIdx.x * blockDim.x + threadIdx.x;
    if (i < LAYERS * HDIM) {
        float v = hidden[i];
        hF32[i] = v;
        hF16[i] = (_Float16)v;
    }
    if (i == 0) *word = *sos;
}

// initial embedding gather (step 0 only; later steps fuse it into logp write)
__global__ void dec_embed(const float* __restrict__ emb, const int* __restrict__ word,
                          _Float16* __restrict__ x) {
    int i = blockIdx.x * blockDim.x + threadIdx.x;
    int w = *word;
    if (i < HDIM) x[i] = (_Float16)emb[(long)w * HDIM + i];
}

// =====================================================================
// WMMA matvec tile: 16 rows of A[rows x HDIM] (f16, row major) times x,
// x replicated across all 16 B columns -> every D column = the matvec.
// A VGPR layout (16-bit A 16x32): lane m=lane&15 is row M; khalf=lane>>4
// picks K sub-blocks {0..7,16..23} vs {8..15,24..31}.
// B VGPR layout (16-bit B 32x16): same K order as A, lanes 0-15 hold
// K={0..7,16..23}, lanes 16-31 hold K={8..15,24..31} (all columns identical here).
// =====================================================================
__device__ __forceinline__ v8f dec_mv_tile(const _Float16* __restrict__ A,
                                           const _Float16* __restrict__ x,
                                           int row0, int lane) {
    const int m     = lane & 15;
    const int khalf = lane >> 4;
    const _Float16* arow = A + (size_t)(row0 + m) * HDIM;
    v8f c = {};
    for (int kk = 0; kk < HDIM; kk += 32) {
        v8h a0 = *(const v8h*)(arow + kk + khalf * 8);        // K = kk+khalf*8 .. +7
        v8h a1 = *(const v8h*)(arow + kk + 16 + khalf * 8);   // K = kk+16+khalf*8 .. +7
        v8h b0 = *(const v8h*)(x + kk + khalf * 8);           // K = kk+khalf*8 .. +7
        v8h b1 = *(const v8h*)(x + kk + 16 + khalf * 8);      // K = kk+16+khalf*8 .. +7
        v16h a, b;
#pragma unroll
        for (int e = 0; e < 8; ++e) {
            a[e] = a0[e]; a[e + 8] = a1[e];
            b[e] = b0[e]; b[e + 8] = b1[e];
        }
        c = __builtin_amdgcn_wmma_f32_16x16x32_f16(false, a, false, b,
                                                   (short)0, c, false, false);
    }
    return c;
}

// C/D layout: VGPR r, lanes 0-15 -> M=r, lanes 16-31 -> M=8+r. All columns
// equal, so lane 0 carries rows 0..7, lane 16 carries rows 8..15.
__device__ __forceinline__ void dec_store_tile(v8f c, int row0, int lane,
                                               const float* __restrict__ bias,
                                               float* __restrict__ y, int rows_valid) {
    if (lane == 0 || lane == 16) {
        const int mbase = row0 + ((lane == 16) ? 8 : 0);
#pragma unroll
        for (int r = 0; r < 8; ++r) {
            const int row = mbase + r;
            if (row < rows_valid) y[row] = c[r] + bias[row];
        }
    }
}

// GRU pre-activations for one layer: waves 0..191 compute gi = w_ih@x + b_ih,
// waves 192..383 compute gh = w_hh@h + b_hh. (48 blocks x 8 waves)
__global__ void dec_gru_mv(const _Float16* __restrict__ wih, const _Float16* __restrict__ whh,
                           const _Float16* __restrict__ xv,  const _Float16* __restrict__ hv,
                           const float* __restrict__ bih,    const float* __restrict__ bhh,
                           float* __restrict__ gi,           float* __restrict__ gh) {
    const int wave = blockIdx.x * (blockDim.x >> 5) + (threadIdx.x >> 5);
    const int lane = threadIdx.x & 31;
    const bool isH = wave >= (G3H / 16);
    const int tile = isH ? wave - (G3H / 16) : wave;
    const _Float16* A  = isH ? whh : wih;
    const _Float16* bx = isH ? hv  : xv;
    const float* bias  = isH ? bhh : bih;
    float* y           = isH ? gh  : gi;
    const int row0 = tile * 16;
    v8f c = dec_mv_tile(A, bx, row0, lane);
    dec_store_tile(c, row0, lane, bias, y, G3H);
}

// Elementwise GRU gates (torch order r,z,n): h' = (1-z)*n + z*h
__global__ void dec_gru_gate(const float* __restrict__ gi, const float* __restrict__ gh,
                             float* __restrict__ hF32, _Float16* __restrict__ hF16,
                             _Float16* __restrict__ xv) {
    int i = blockIdx.x * blockDim.x + threadIdx.x;
    if (i >= HDIM) return;
    float r = 1.0f / (1.0f + expf(-(gi[i] + gh[i])));
    float z = 1.0f / (1.0f + expf(-(gi[HDIM + i] + gh[HDIM + i])));
    float n = tanhf(gi[2 * HDIM + i] + r * gh[2 * HDIM + i]);
    float hn = (1.0f - z) * n + z * hF32[i];
    hF32[i] = hn;
    _Float16 hh = (_Float16)hn;
    hF16[i] = hh;   // next step's w_hh operand
    xv[i]   = hh;   // input to next layer / logits
}

// =====================================================================
// Fused logits matvec + per-block softmax stats.
// Each block: 8 waves x 16 rows = 128 logits, held in lanes 0/16 after
// WMMA. Push through LDS, write logits, and emit one partial
// (max, first-argmax, sum exp(v-max)) per block. Slot order == row order
// so first-index tie-breaking matches jnp.argmax.
// =====================================================================
__global__ void dec_logits_mv_stats(const _Float16* __restrict__ W, const _Float16* __restrict__ x,
                                    const float* __restrict__ bias, float* __restrict__ y,
                                    float* __restrict__ pmax, int* __restrict__ pidx,
                                    float* __restrict__ psum) {
    __shared__ float vals[128];
    __shared__ float rmax[128];
    __shared__ int   ridx[128];
    __shared__ float rsum[128];
    const int wave = threadIdx.x >> 5;
    const int lane = threadIdx.x & 31;
    const int row0 = (blockIdx.x * 8 + wave) * 16;
    v8f c = dec_mv_tile(W, x, row0, lane);
    if (lane == 0 || lane == 16) {
        const int half = (lane == 16) ? 8 : 0;
#pragma unroll
        for (int r = 0; r < 8; ++r) {
            const int row = row0 + half + r;
            float v = NEG_INF;
            if (row < VOCAB) { v = c[r] + bias[row]; y[row] = v; }
            vals[wave * 16 + half + r] = v;
        }
    }
    __syncthreads();
    const int tid = threadIdx.x;
    if (tid < 128) { rmax[tid] = vals[tid]; ridx[tid] = blockIdx.x * 128 + tid; }
    __syncthreads();
    for (int s = 64; s > 0; s >>= 1) {
        if (tid < s) {
            float ov = rmax[tid + s]; int oi = ridx[tid + s];
            if (ov > rmax[tid] || (ov == rmax[tid] && oi < ridx[tid])) {
                rmax[tid] = ov; ridx[tid] = oi;
            }
        }
        __syncthreads();
    }
    float m = rmax[0];
    if (tid < 128) rsum[tid] = expf(vals[tid] - m);   // exp(-inf)=0 for pad rows
    __syncthreads();
    for (int s = 64; s > 0; s >>= 1) {
        if (tid < s) rsum[tid] += rsum[tid + s];
        __syncthreads();
    }
    if (tid == 0) { pmax[blockIdx.x] = m; pidx[blockIdx.x] = ridx[0]; psum[blockIdx.x] = rsum[0]; }
}

// Final reduce over 393 partials (512 threads): global max/argmax,
// logZ = M + log(sum s_b*exp(m_b-M)); publish next token + token output.
__global__ void dec_smax_reduce(const float* __restrict__ pmax, const int* __restrict__ pidx,
                                const float* __restrict__ psum,
                                int* __restrict__ word, float* __restrict__ logZ,
                                float* __restrict__ tok_out) {
    __shared__ float smax[512];
    __shared__ int   sidx[512];
    __shared__ float ssum[512];
    const int tid = threadIdx.x;
    float m = (tid < NBLK_P) ? pmax[tid] : NEG_INF;
    int  id = (tid < NBLK_P) ? pidx[tid] : 0x7fffffff;
    smax[tid] = m; sidx[tid] = id;
    __syncthreads();
    for (int s = 256; s > 0; s >>= 1) {
        if (tid < s) {
            float ov = smax[tid + s]; int oi = sidx[tid + s];
            if (ov > smax[tid] || (ov == smax[tid] && oi < sidx[tid])) {
                smax[tid] = ov; sidx[tid] = oi;
            }
        }
        __syncthreads();
    }
    float M = smax[0]; int I = sidx[0];
    ssum[tid] = (tid < NBLK_P) ? psum[tid] * expf(m - M) : 0.0f;
    __syncthreads();
    for (int s = 256; s > 0; s >>= 1) {
        if (tid < s) ssum[tid] += ssum[tid + s];
        __syncthreads();
    }
    if (tid == 0) {
        *word = I;
        *logZ = M + logf(ssum[0]);
        *tok_out = (float)I;
    }
}

// Fused: write log-probs for this step AND gather next step's embedding
// (word was just updated by dec_smax_reduce; blocks 0..3 cover i<HDIM).
__global__ void dec_logp_embed(const float* __restrict__ logits, const float* __restrict__ logZ,
                               float* __restrict__ out,
                               const float* __restrict__ emb, const int* __restrict__ word,
                               _Float16* __restrict__ x) {
    int i = blockIdx.x * blockDim.x + threadIdx.x;
    if (i < VOCAB) out[i] = logits[i] - *logZ;
    if (i < HDIM)  x[i] = (_Float16)emb[(long)(*word) * HDIM + i];
}

// =====================================================================
// Launch: per-call f16 conversions (deterministic, rewrites all of ws),
// then 128 sequential decode steps, each a chain of 7 kernels.
// =====================================================================
extern "C" void kernel_launch(void* const* d_in, const int* in_sizes, int n_in,
                              void* d_out, int out_size, void* d_ws, size_t ws_size,
                              hipStream_t stream) {
    (void)in_sizes; (void)n_in; (void)out_size; (void)ws_size;
    const float* hidden = (const float*)d_in[0];  // [L,H]
    const float* emb    = (const float*)d_in[1];  // [V,H]
    const float* w_ih   = (const float*)d_in[2];  // [L,3H,H]
    const float* w_hh   = (const float*)d_in[3];  // [L,3H,H]
    const float* b_ih   = (const float*)d_in[4];  // [L,3H]
    const float* b_hh   = (const float*)d_in[5];  // [L,3H]
    const float* out_w  = (const float*)d_in[6];  // [V,H]
    const float* out_b  = (const float*)d_in[7];  // [V]
    const int*   sos    = (const int*)d_in[8];    // scalar
    // d_in[9] = bound (compile-time NSTEPS)

    // ---- workspace layout (all offsets 256B-aligned) ----
    char* ws = (char*)d_ws;
    const size_t SZ_OUTW = (size_t)VPAD * HDIM * 2;          // 103.0 MB
    const size_t SZ_W    = (size_t)LAYERS * G3H * HDIM * 2;  // 12.6 MB each
    size_t off = 0;
    _Float16* outwF16 = (_Float16*)(ws + off); off += SZ_OUTW;
    _Float16* wihF16  = (_Float16*)(ws + off); off += SZ_W;
    _Float16* whhF16  = (_Float16*)(ws + off); off += SZ_W;
    _Float16* xF16    = (_Float16*)(ws + off); off += 4096;
    _Float16* hF16    = (_Float16*)(ws + off); off += 4096;   // [L,H]
    float*    hF32    = (float*)   (ws + off); off += 8192;   // [L,H]
    float*    gi      = (float*)   (ws + off); off += 12288;  // [3H]
    float*    gh      = (float*)   (ws + off); off += 12288;  // [3H]
    float*    logits  = (float*)   (ws + off); off += (size_t)VPAD * 4;
    float*    pmax    = (float*)   (ws + off); off += 2048;   // NBLK_P=393
    int*      pidx    = (int*)     (ws + off); off += 2048;
    float*    psum    = (float*)   (ws + off); off += 2048;
    int*      word    = (int*)     (ws + off); off += 256;
    float*    logZ    = (float*)   (ws + off); off += 256;

    float* out_logp = (float*)d_out;                       // [NSTEPS, VOCAB]
    float* out_tok  = out_logp + (size_t)NSTEPS * VOCAB;   // [NSTEPS]

    // ---- one-time (per call) weight conversion to f16 (L2-resident set) ----
    {
        long nt = (long)VPAD * HDIM;
        int blocks = (int)((nt / 4 + 255) / 256);
        dec_cvt_f16<<<blocks, 256, 0, stream>>>(out_w, outwF16, (long)VOCAB * HDIM, nt);
    }
    {
        long nt = (long)LAYERS * G3H * HDIM;
        int blocks = (int)((nt / 4 + 255) / 256);
        dec_cvt_f16<<<blocks, 256, 0, stream>>>(w_ih, wihF16, nt, nt);
        dec_cvt_f16<<<blocks, 256, 0, stream>>>(w_hh, whhF16, nt, nt);
    }
    dec_init_state<<<(LAYERS * HDIM + 255) / 256, 256, 0, stream>>>(hidden, hF32, hF16, sos, word);
    dec_embed<<<(HDIM + 255) / 256, 256, 0, stream>>>(emb, word, xF16);

    // ---- sequential greedy decode: 7 kernels per step ----
    for (int t = 0; t < NSTEPS; ++t) {
        for (int l = 0; l < LAYERS; ++l) {
            dec_gru_mv<<<(2 * G3H / 16) / 8, 256, 0, stream>>>(
                wihF16 + (size_t)l * G3H * HDIM, whhF16 + (size_t)l * G3H * HDIM,
                xF16, hF16 + (size_t)l * HDIM,
                b_ih + (size_t)l * G3H, b_hh + (size_t)l * G3H, gi, gh);
            dec_gru_gate<<<(HDIM + 255) / 256, 256, 0, stream>>>(
                gi, gh, hF32 + (size_t)l * HDIM, hF16 + (size_t)l * HDIM, xF16);
        }
        dec_logits_mv_stats<<<NBLK_P, 256, 0, stream>>>(outwF16, xF16, out_b, logits,
                                                        pmax, pidx, psum);
        dec_smax_reduce<<<1, 512, 0, stream>>>(pmax, pidx, psum, word, logZ, out_tok + t);
        // writes step-t log-probs and gathers x=emb[word] for step t+1
        dec_logp_embed<<<NBLK_V, 256, 0, stream>>>(logits, logZ, out_logp + (size_t)t * VOCAB,
                                                   emb, word, xF16);
    }
}
